// MultimodalCrossAttention_76244259439358
// MI455X (gfx1250) — compile-verified, matrix operand order checked by reading, not run on hardware
//
#include <hip/hip_runtime.h>
#include <hip/hip_bf16.h>
#include <math.h>

// ---------------------------------------------------------------------------
// Types for CDNA5 (gfx1250) WMMA, wave32.
// ---------------------------------------------------------------------------
typedef __attribute__((ext_vector_type(16))) _Float16 v16h;
typedef __attribute__((ext_vector_type(8)))  float    v8f;
typedef __attribute__((ext_vector_type(4)))  unsigned u32x4;

union V16H {
  u32x4    q[2];   // two 16-byte halves (halfwords 0..7 / 8..15)
  v16h     v;
  _Float16 h[16];
};

__device__ __forceinline__ v8f wmma_f16(const V16H& a, const V16H& b, v8f c) {
  // D = A(16x32 f16) x B(32x16 f16) + C(16x16 f32)
  return __builtin_amdgcn_wmma_f32_16x16x32_f16(
      /*neg_a=*/false, a.v, /*neg_b=*/false, b.v,
      /*c_mod=*/(short)0, c, /*reuse_a=*/false, /*reuse_b=*/false);
}

// Async DMA: global -> LDS, 16 bytes per lane, tracked by ASYNCcnt.
__device__ __forceinline__ void async_ld_b128(unsigned lds_off_bytes,
                                              const _Float16* g) {
  asm volatile("global_load_async_to_lds_b128 %0, %1, off"
               :: "v"(lds_off_bytes), "v"(g)
               : "memory");
}

// ---------------------------------------------------------------------------
// f32 -> f16 conversion
// ---------------------------------------------------------------------------
__global__ __launch_bounds__(256) void cvt_f32_to_f16(
    const float* __restrict__ in, _Float16* __restrict__ out, int n) {
  int i = blockIdx.x * blockDim.x + threadIdx.x;
  if (i < n) out[i] = (_Float16)in[i];
}

// ---------------------------------------------------------------------------
// GEMM: Y = X @ W^T.  X:[T,K] f16 row-major, W:[N,K] f16 row-major.
// Wave computes C^T tile: A = 4 W feature-tiles (M=feature), B = X image
// (N=16 tokens).  C layout: lane = token, vgpr = feature -> contiguous stores.
// MODE 0: f16 out [T][N].  MODE 1: f32 out [T][N] + bias.
// ---------------------------------------------------------------------------
template <int MODE>
__global__ __launch_bounds__(256) void gemm_xwT(
    const _Float16* __restrict__ X, const _Float16* __restrict__ W,
    void* __restrict__ outp, const float* __restrict__ bias,
    int T, int N, int K) {
  const int lane  = threadIdx.x & 31;
  const int wave  = (int)((blockIdx.x * (size_t)blockDim.x + threadIdx.x) >> 5);
  const int nGrp  = N >> 6;                 // feature groups of 64
  const int tTile = wave / nGrp;            // token tile (16 tokens)
  const int fGrp  = wave - tTile * nGrp;    // feature group
  if (tTile * 16 >= T) return;
  const int ln    = lane & 15;
  const int baseK = (lane >= 16) ? 8 : 0;   // ISA 16-bit A/B-operand K split
  const int tok   = tTile * 16 + ln;

  const _Float16* xrow = X + (size_t)tok * K;
  const _Float16* wrow[4];
#pragma unroll
  for (int t = 0; t < 4; ++t)
    wrow[t] = W + (size_t)(fGrp * 64 + t * 16 + ln) * K;

  v8f acc[4] = {v8f{0}, v8f{0}, v8f{0}, v8f{0}};
  for (int k0 = 0; k0 < K; k0 += 32) {
    V16H b;  // X image: serves as B operand (X^T)
    b.q[0] = *(const u32x4*)(xrow + k0 + baseK);
    b.q[1] = *(const u32x4*)(xrow + k0 + 16 + baseK);
#pragma unroll
    for (int t = 0; t < 4; ++t) {
      V16H a;  // W rows as A operand
      a.q[0] = *(const u32x4*)(wrow[t] + k0 + baseK);
      a.q[1] = *(const u32x4*)(wrow[t] + k0 + 16 + baseK);
      acc[t] = wmma_f16(a, b, acc[t]);
    }
  }

  const int hi8 = (lane >= 16) ? 8 : 0;     // C layout: M = vgpr + hi8
  if (MODE == 0) {
    _Float16* out = (_Float16*)outp;
#pragma unroll
    for (int t = 0; t < 4; ++t) {
      V16H pk;
#pragma unroll
      for (int r = 0; r < 8; ++r) pk.h[r] = (_Float16)acc[t][r];
      *(u32x4*)(out + (size_t)tok * N + fGrp * 64 + t * 16 + hi8) = pk.q[0];
    }
  } else {
    float* out = (float*)outp;
#pragma unroll
    for (int t = 0; t < 4; ++t) {
      const int fb = fGrp * 64 + t * 16 + hi8;
      float* o = out + (size_t)tok * N + fb;
#pragma unroll
      for (int r = 0; r < 8; ++r) o[r] = acc[t][r] + bias[fb + r];
    }
  }
}

// ---------------------------------------------------------------------------
// RoPE (interleaved pairs) + relayout [B,S,H*64] -> [B,H,S,64], f16.
// `scale` folds the attention 1/sqrt(hd) into Q so the flash loop never
// rescales scores.
// ---------------------------------------------------------------------------
__global__ __launch_bounds__(256) void rope_relayout(
    const _Float16* __restrict__ P, _Float16* __restrict__ outR,
    int B_, int H_, int S, float scale) {
  long idx   = blockIdx.x * (long)blockDim.x + threadIdx.x;
  long total = (long)B_ * S * H_ * 32;
  if (idx >= total) return;
  int  i  = (int)(idx & 31);           // pair index within head (0..31)
  long t2 = idx >> 5;
  int  h  = (int)(t2 % H_);
  long t3 = t2 / H_;
  int  s  = (int)(t3 % S);
  int  b  = (int)(t3 / S);

  const _Float16* src = P + ((size_t)b * S + s) * (H_ * 64) + h * 64 + 2 * i;
  float x1 = (float)src[0], x2 = (float)src[1];
  float inv = __powf(10000.0f, -(float)(2 * i) / 64.0f);
  float sn, cs;
  __sincosf((float)s * inv, &sn, &cs);
  _Float16* dst = outR + (((size_t)(b * H_ + h)) * S + s) * 64 + 2 * i;
  dst[0] = (_Float16)((x1 * cs - x2 * sn) * scale);
  dst[1] = (_Float16)((x1 * sn + x2 * cs) * scale);
}

// ---------------------------------------------------------------------------
// V relayout+transpose: [B,S,H*64] -> [B,H,64,S] f16.
// ---------------------------------------------------------------------------
__global__ __launch_bounds__(256) void v_transpose(
    const _Float16* __restrict__ P, _Float16* __restrict__ Vt,
    int B_, int H_, int S) {
  long idx   = blockIdx.x * (long)blockDim.x + threadIdx.x;
  long total = (long)B_ * H_ * 64 * S;
  if (idx >= total) return;
  int  s  = (int)(idx % S);
  long t2 = idx / S;
  int  d  = (int)(t2 & 63);
  int  bh = (int)(t2 >> 6);
  int  b  = bh / H_, h = bh % H_;
  Vt[idx] = P[((size_t)b * S + s) * (H_ * 64) + h * 64 + d];
}

// ---------------------------------------------------------------------------
// Flash attention. Block = 8 waves sharing one (b,h); each wave owns a 16-row
// q tile. K/V 32-kv chunks are staged global->LDS with async b128 DMA
// (double-buffered, ASYNCcnt), then consumed via ds_load_b128 as WMMA
// A-operands:
//   S^T = K_tile x Q^T      (C layout: M=kv across vgprs, N=q across lanes)
//   -> per-lane softmax for q = lane&15 (one shfl_xor(16) combine)
//   -> P^T C-layout repacks IN-LANE into the P A-operand image
//   O^T += V^T_tile x P^T   (C layout: N=q per lane -> per-lane rescale)
// All K operands are loaded before the score WMMAs and the V operands are
// prefetched before the softmax so LDS latency hides under VALU work.
// LDS rows padded (64->72, 32->40 halves) to spread the 64 banks.
// ---------------------------------------------------------------------------
#define KROW 72   // padded K-chunk row stride (halves)
#define VROW 40   // padded V-chunk row stride (halves)

__global__ __launch_bounds__(256) void attn_fwd(
    const _Float16* __restrict__ Qr,   // [B*H, Sx, 64]  (pre-scaled by 1/8)
    const _Float16* __restrict__ Kr,   // [B*H, Sc, 64]
    const _Float16* __restrict__ Vt,   // [B*H, 64, Sc]
    _Float16* __restrict__ Obuf,       // [B*Sx, H*64]
    int B_, int H_, int Sx, int Sc) {
  __shared__ alignas(16) _Float16 sK[2][32 * KROW];  // 2 x 4.5 KB
  __shared__ alignas(16) _Float16 sV[2][64 * VROW];  // 2 x 5.0 KB

  const int tid   = threadIdx.x;
  const int lane  = tid & 31;
  const int wv    = tid >> 5;                  // 0..7 waves per block
  const int qgPer = Sx / 128;                  // q-groups (8 tiles) per (b,h)
  const int qg    = blockIdx.x % qgPer;
  const int bh    = blockIdx.x / qgPer;
  const int qt    = qg * 8 + wv;               // 16-row q tile index
  const int ln    = lane & 15;
  const int baseK = (lane >= 16) ? 8 : 0;

  const _Float16* qrow  = Qr + ((size_t)bh * Sx + qt * 16 + ln) * 64;
  const _Float16* kbase = Kr + (size_t)bh * Sc * 64;
  const _Float16* vbase = Vt + (size_t)bh * 64 * Sc;

  // Per-thread slice of the cooperative chunk DMA (2 x b128 per thread).
  const int kr = tid >> 3, kc = (tid & 7) * 8;  // K: 32 rows x 64 halves
  const int vd = tid >> 2, vj = (tid & 3) * 8;  // V: 64 rows x 32 halves

  // Q tile image (row-major rows => A-image => reusable as B operand Q^T)
  V16H qimg[2];
  qimg[0].q[0] = *(const u32x4*)(qrow + baseK);
  qimg[0].q[1] = *(const u32x4*)(qrow + 16 + baseK);
  qimg[1].q[0] = *(const u32x4*)(qrow + 32 + baseK);
  qimg[1].q[1] = *(const u32x4*)(qrow + 48 + baseK);

  v8f   o[4] = {v8f{0}, v8f{0}, v8f{0}, v8f{0}};
  float m = -3.0e38f, l = 0.0f;

  // Prologue: DMA chunk 0 into buffer 0.
  async_ld_b128((unsigned)(uintptr_t)&sK[0][kr * KROW + kc],
                kbase + (size_t)kr * 64 + kc);
  async_ld_b128((unsigned)(uintptr_t)&sV[0][vd * VROW + vj],
                vbase + (size_t)vd * Sc + vj);

  const int nC = Sc / 32;
  for (int c = 0; c < nC; ++c) {
    const int buf = c & 1;
    if (c + 1 < nC) {
      const int kv1 = (c + 1) * 32;
      async_ld_b128((unsigned)(uintptr_t)&sK[buf ^ 1][kr * KROW + kc],
                    kbase + (size_t)(kv1 + kr) * 64 + kc);
      async_ld_b128((unsigned)(uintptr_t)&sV[buf ^ 1][vd * VROW + vj],
                    vbase + (size_t)vd * Sc + kv1 + vj);
      asm volatile("s_wait_asynccnt 0x2" ::: "memory");  // chunk c resident
    } else {
      asm volatile("s_wait_asynccnt 0x0" ::: "memory");
    }
    __syncthreads();  // all waves' DMA for chunk c visible

    const _Float16* sKb = sK[buf];
    const _Float16* sVb = sV[buf];

    // ---- load all K operands, then prefetch V operands -------------------
    const _Float16* k0p = sKb + (size_t)ln * KROW;
    const _Float16* k1p = sKb + (size_t)(16 + ln) * KROW;
    V16H ka[4];
    ka[0].q[0] = *(const u32x4*)(k0p + baseK);
    ka[0].q[1] = *(const u32x4*)(k0p + 16 + baseK);
    ka[1].q[0] = *(const u32x4*)(k0p + 32 + baseK);
    ka[1].q[1] = *(const u32x4*)(k0p + 48 + baseK);
    ka[2].q[0] = *(const u32x4*)(k1p + baseK);
    ka[2].q[1] = *(const u32x4*)(k1p + 16 + baseK);
    ka[3].q[0] = *(const u32x4*)(k1p + 32 + baseK);
    ka[3].q[1] = *(const u32x4*)(k1p + 48 + baseK);
    V16H va[4];
#pragma unroll
    for (int t = 0; t < 4; ++t) {
      const _Float16* vr = sVb + (size_t)(t * 16 + ln) * VROW;
      va[t].q[0] = *(const u32x4*)(vr + baseK);
      va[t].q[1] = *(const u32x4*)(vr + 16 + baseK);
    }

    // ---- scores: two 16x16 S^T tiles (kv rows 0..15 / 16..31 of chunk) ---
    v8f s0 = v8f{0}, s1 = v8f{0};
    s0 = wmma_f16(ka[0], qimg[0], s0);
    s0 = wmma_f16(ka[1], qimg[1], s0);
    s1 = wmma_f16(ka[2], qimg[0], s1);
    s1 = wmma_f16(ka[3], qimg[1], s1);

    // ---- online softmax (per-lane q = lane&15; partner lane = lane^16) ---
    float cmax = -3.0e38f;
#pragma unroll
    for (int r = 0; r < 8; ++r)
      cmax = fmaxf(cmax, fmaxf(s0[r], s1[r]));
    cmax = fmaxf(cmax, __shfl_xor(cmax, 16, 32));
    float mn    = fmaxf(m, cmax);
    float alpha = __expf(m - mn);
    float psum  = 0.0f;
    V16H  p;  // P A-operand image: halfwords 0..7 <- tile0, 8..15 <- tile1
#pragma unroll
    for (int r = 0; r < 8; ++r) {
      float p0 = __expf(s0[r] - mn);
      float p1 = __expf(s1[r] - mn);
      psum += p0 + p1;
      p.h[r]     = (_Float16)p0;
      p.h[8 + r] = (_Float16)p1;
    }
    psum += __shfl_xor(psum, 16, 32);
    l = l * alpha + psum;
    m = mn;

#pragma unroll
    for (int t = 0; t < 4; ++t)
#pragma unroll
      for (int r = 0; r < 8; ++r) o[t][r] *= alpha;

    // ---- O^T += V^T x P^T -----------------------------------------------
#pragma unroll
    for (int t = 0; t < 4; ++t) o[t] = wmma_f16(va[t], p, o[t]);

    __syncthreads();  // done reading buf before it is refilled
  }

  // ---- normalize + store: O^T lane = token, vgprs = contiguous features --
  const float inv = 1.0f / l;
  const int   b_  = bh / H_;
  const int   h_  = bh % H_;
  const int   hi8 = (lane >= 16) ? 8 : 0;
  const size_t row = (size_t)b_ * Sx + qt * 16 + ln;
#pragma unroll
  for (int t = 0; t < 4; ++t) {
    V16H pk;
#pragma unroll
    for (int r = 0; r < 8; ++r) pk.h[r] = (_Float16)(o[t][r] * inv);
    *(u32x4*)(Obuf + row * (H_ * 64) + h_ * 64 + t * 16 + hi8) = pk.q[0];
  }
}

// ---------------------------------------------------------------------------
// Host-side launch
// ---------------------------------------------------------------------------
extern "C" void kernel_launch(void* const* d_in, const int* in_sizes, int n_in,
                              void* d_out, int out_size, void* d_ws,
                              size_t ws_size, hipStream_t stream) {
  (void)n_in; (void)out_size; (void)ws_size;
  const float* x   = (const float*)d_in[0];
  const float* ctx = (const float*)d_in[1];
  const float* Wq  = (const float*)d_in[2];
  const float* Wv  = (const float*)d_in[3];
  const float* Wo  = (const float*)d_in[4];
  const float* bo  = (const float*)d_in[5];

  const int D = 1024, H = 16, Sx = 2048, Sc = 2048;
  const int B  = in_sizes[0] / (Sx * D);
  const int Tq = B * Sx;
  const int Tk = B * Sc;

  _Float16* ws = (_Float16*)d_ws;
  size_t off = 0;
  _Float16* xh  = ws + off; off += (size_t)Tq * D;  // x f16
  _Float16* ch  = ws + off; off += (size_t)Tk * D;  // context f16
  _Float16* wqh = ws + off; off += (size_t)D * D;
  _Float16* wvh = ws + off; off += (size_t)D * D;
  _Float16* woh = ws + off; off += (size_t)D * D;
  _Float16* Qt  = ws + off; off += (size_t)Tq * D;  // q pre-rope [T][D]
  _Float16* Kt  = ws + off; off += (size_t)Tk * D;  // k pre-rope
  _Float16* Vt0 = ws + off; off += (size_t)Tk * D;  // v token-major
  _Float16* Qr  = ws + off; off += (size_t)Tq * D;  // [B,H,Sx,64]
  _Float16* Kr  = ws + off; off += (size_t)Tk * D;  // [B,H,Sc,64]
  _Float16* Vtr = ws + off; off += (size_t)Tk * D;  // [B,H,64,Sc]
  _Float16* Ob  = ws + off; off += (size_t)Tq * D;  // attention out [T][D]

  // 1) convert inputs/weights to f16
  auto cvt = [&](const float* a, _Float16* b, int n) {
    cvt_f32_to_f16<<<(n + 255) / 256, 256, 0, stream>>>(a, b, n);
  };
  cvt(x, xh, Tq * D);
  cvt(ctx, ch, Tk * D);
  cvt(Wq, wqh, D * D);
  cvt(Wv, wvh, D * D);
  cvt(Wo, woh, D * D);

  // 2) projections (WMMA GEMMs): q = x Wq^T, k = ctx Wq^T, v = ctx Wv^T
  const int blkQ = (Tq / 16) * (D / 64) / 8;  // 8 waves/block
  const int blkK = (Tk / 16) * (D / 64) / 8;
  gemm_xwT<0><<<blkQ, 256, 0, stream>>>(xh, wqh, Qt, nullptr, Tq, D, D);
  gemm_xwT<0><<<blkK, 256, 0, stream>>>(ch, wqh, Kt, nullptr, Tk, D, D);
  gemm_xwT<0><<<blkK, 256, 0, stream>>>(ch, wvh, Vt0, nullptr, Tk, D, D);

  // 3) RoPE + head relayout (Q folds 1/sqrt(64)); V transpose
  long nrq = (long)B * Sx * H * 32;
  long nrk = (long)B * Sc * H * 32;
  rope_relayout<<<(int)((nrq + 255) / 256), 256, 0, stream>>>(Qt, Qr, B, H, Sx,
                                                              0.125f);
  rope_relayout<<<(int)((nrk + 255) / 256), 256, 0, stream>>>(Kt, Kr, B, H, Sc,
                                                              1.0f);
  long nv = (long)B * H * 64 * Sc;
  v_transpose<<<(int)((nv + 255) / 256), 256, 0, stream>>>(Vt0, Vtr, B, H, Sc);

  // 4) flash attention (8 q-tiles per block, async K/V staging into LDS)
  attn_fwd<<<B * H * (Sx / 128), 256, 0, stream>>>(Qr, Kr, Vtr, Ob, B, H, Sx, Sc);

  // 5) output projection + bias -> f32 d_out
  gemm_xwT<1><<<blkQ, 256, 0, stream>>>(Ob, woh, d_out, bo, Tq, D, D);
}
